// Space_Attention_67577015435619
// MI455X (gfx1250) — matrix operand order repaired and checked by reading, not compiled, not run on hardware
//
#include <hip/hip_runtime.h>
#include <hip/hip_bf16.h>
#include <math.h>

// Problem constants (match reference).
#define IN_DIM  1024
#define OUT_DIM 1024
#define CENTERS 32
#define NROWS   4096

typedef __attribute__((ext_vector_type(16))) __bf16 v16bf;
typedef __attribute__((ext_vector_type(8)))  __bf16 v8bf;
typedef __attribute__((ext_vector_type(8)))  float  v8f;

__device__ __forceinline__ unsigned short f2bf(float f) {
  // round-to-nearest-even f32 -> bf16
  unsigned u = __float_as_uint(f);
  u += 0x7FFFu + ((u >> 16) & 1u);
  return (unsigned short)(u >> 16);
}

__device__ __forceinline__ v8f zero8() {
  v8f z = {0.f, 0.f, 0.f, 0.f, 0.f, 0.f, 0.f, 0.f};
  return z;
}

// CDNA5 async global->LDS copy (ASYNCcnt-tracked, lane-addressed LDS dest).
__device__ __forceinline__ void async_copy_b128(unsigned lds_byte_addr,
                                                unsigned gbl_byte_off,
                                                const void* base) {
  asm volatile("global_load_async_to_lds_b128 %0, %1, %2"
               :: "v"(lds_byte_addr), "v"(gbl_byte_off), "s"(base)
               : "memory");
}
__device__ __forceinline__ void wait_async0() {
  asm volatile("s_wait_asynccnt 0x0" ::: "memory");
}

// ---------------------------------------------------------------------------
// Kernel 1: inverse L2 norms of k_c columns (32 values). One wave.
// ---------------------------------------------------------------------------
__global__ void knorm_kernel(const float* __restrict__ kc, float* __restrict__ invkn) {
  int c = threadIdx.x;  // 32 threads
  float s = 0.f;
  for (int i = 0; i < IN_DIM; ++i) {
    float v = kc[i * CENTERS + c];
    s = fmaf(v, v, s);
  }
  invkn[c] = 1.0f / fmaxf(sqrtf(s), 1e-12f);
}

// ---------------------------------------------------------------------------
// Kernel 2: attn[n, c] = softmax_c( (1/32) * <x_n/||x_n||, k_c/||k_c||> ).
// One wave32 per row; lane c owns center c; cross-lane softmax via shfl_xor.
// ---------------------------------------------------------------------------
__global__ __launch_bounds__(128) void attn_kernel(const float* __restrict__ x,
                                                   const float* __restrict__ kc,
                                                   const float* __restrict__ invkn,
                                                   float* __restrict__ attn) {
  const int lane = threadIdx.x & 31;
  const int n    = blockIdx.x * 4 + (threadIdx.x >> 5);
  const float* xr = x + (size_t)n * IN_DIM;

  float xsq = 0.f;
#pragma unroll 4
  for (int t = 0; t < 32; ++t) {
    float v = xr[lane + t * 32];
    xsq = fmaf(v, v, xsq);
  }
#pragma unroll
  for (int o = 16; o > 0; o >>= 1) xsq += __shfl_xor(xsq, o, 32);

  float dot = 0.f;
  for (int i = 0; i < IN_DIM; i += 4) {
    float4 xv = *(const float4*)(xr + i);
    dot = fmaf(xv.x, kc[(i + 0) * CENTERS + lane], dot);
    dot = fmaf(xv.y, kc[(i + 1) * CENTERS + lane], dot);
    dot = fmaf(xv.z, kc[(i + 2) * CENTERS + lane], dot);
    dot = fmaf(xv.w, kc[(i + 3) * CENTERS + lane], dot);
  }

  float invx  = 1.0f / fmaxf(sqrtf(xsq), 1e-12f);
  float logit = 0.03125f * dot * invx * invkn[lane];  // scale = 1/sqrt(1024)

  float m = logit;
#pragma unroll
  for (int o = 16; o > 0; o >>= 1) m = fmaxf(m, __shfl_xor(m, o, 32));
  float e = __expf(logit - m);
  float s = e;
#pragma unroll
  for (int o = 16; o > 0; o >>= 1) s += __shfl_xor(s, o, 32);

  attn[n * CENTERS + lane] = e / s;
}

// ---------------------------------------------------------------------------
// Kernel 2.5 (fast path): v_c f32 [c][k][n]  ->  vT bf16 [c][n][k]
// Tiled 64x64 transpose through LDS; one-time 128MB read / 64MB write
// (~8us at 23.3 TB/s, amortized over the compute-bound GEMM).
// ---------------------------------------------------------------------------
__global__ __launch_bounds__(256) void vconv_kernel(const float* __restrict__ vc,
                                                    unsigned short* __restrict__ vt) {
  __shared__ unsigned short tile[64 * 65];
  const int c  = blockIdx.z;
  const int kb = blockIdx.y * 64;
  const int nb = blockIdx.x * 64;
  const float* src = vc + ((size_t)c << 20);
#pragma unroll
  for (int p = 0; p < 16; ++p) {
    int idx = threadIdx.x + p * 256;
    int kl = idx >> 6, nl = idx & 63;
    tile[kl * 65 + nl] = f2bf(src[(size_t)(kb + kl) * OUT_DIM + nb + nl]);
  }
  __syncthreads();
  unsigned short* dst = vt + ((size_t)c << 20);
#pragma unroll
  for (int p = 0; p < 2; ++p) {
    int q = threadIdx.x + p * 256;     // 512 (n, k8) pairs
    int nl = q >> 3, k8 = (q & 7) * 8;
    unsigned short t[8];
#pragma unroll
    for (int j = 0; j < 8; ++j) t[j] = tile[(k8 + j) * 65 + nl];
    uint4 u;
    u.x = (unsigned)t[0] | ((unsigned)t[1] << 16);
    u.y = (unsigned)t[2] | ((unsigned)t[3] << 16);
    u.z = (unsigned)t[4] | ((unsigned)t[5] << 16);
    u.w = (unsigned)t[6] | ((unsigned)t[7] << 16);
    *(uint4*)(dst + (size_t)(nb + nl) * IN_DIM + kb + k8) = u;
  }
}

// ---------------------------------------------------------------------------
// Shared GEMM geometry: block 128x128 (8 wave32s, 2x4), wave tile 64x32.
// A (x rows, bf16) persistent in LDS; K-transposed, double-buffered B tile.
// ---------------------------------------------------------------------------
#define BM 128
#define BN 128
#define APITCH 1032           // bf16 elements per A row (pad -> conflict-free)
#define BPITCH 40             // bf16 elements per Bt row (32 K + pad 8)
#define SM_A_BYTES (BM * APITCH * 2)                  // 264192
#define SM_B_BYTES (BN * BPITCH * 2)                  // 10240
#define SM_ATTN_OFF (SM_A_BYTES + 2 * SM_B_BYTES)     // 284672
#define SMEM_BYTES (SM_ATTN_OFF + BM * CENTERS * 4)   // 301056 (<=320KB WGP LDS)

__device__ __forceinline__ void stage_a_attn(const float* __restrict__ x,
                                             const float* __restrict__ attn,
                                             __bf16* As, float* atS,
                                             int gm, int tid) {
  for (int idx = tid; idx < BM * (IN_DIM / 4); idx += 256) {
    int m = idx >> 8;
    int q = idx & 255;
    float4 xv = *(const float4*)(x + (size_t)(gm + m) * IN_DIM + q * 4);
    uint2 p;
    p.x = (unsigned)f2bf(xv.x) | ((unsigned)f2bf(xv.y) << 16);
    p.y = (unsigned)f2bf(xv.z) | ((unsigned)f2bf(xv.w) << 16);
    *(uint2*)(As + m * APITCH + q * 4) = p;
  }
  for (int idx = tid; idx < BM * CENTERS; idx += 256) {
    atS[idx] = attn[(size_t)gm * CENTERS + idx];
  }
}

__device__ __forceinline__ void load_a_frags(const __bf16* As, int wm, int lm,
                                             int khalf, int kk, v16bf af[4]) {
  const __bf16* Ab = As + (size_t)(wm + lm) * APITCH + kk * 32 + khalf * 8;
#pragma unroll
  for (int tm = 0; tm < 4; ++tm) {
    v8bf lo = *(const v8bf*)(Ab + tm * 16 * APITCH);
    v8bf hi = *(const v8bf*)(Ab + tm * 16 * APITCH + 16);
    af[tm] = __builtin_shufflevector(lo, hi, 0, 1, 2, 3, 4, 5, 6, 7,
                                     8, 9, 10, 11, 12, 13, 14, 15);
  }
}
__device__ __forceinline__ void load_b_frags(const __bf16* Bp, int wn, int lm,
                                             int khalf, v16bf bfr[2]) {
  const __bf16* Bb = Bp + (size_t)(wn + lm) * BPITCH + khalf * 8;
#pragma unroll
  for (int tn = 0; tn < 2; ++tn) {
    v8bf lo = *(const v8bf*)(Bb + tn * 16 * BPITCH);
    v8bf hi = *(const v8bf*)(Bb + tn * 16 * BPITCH + 16);
    bfr[tn] = __builtin_shufflevector(lo, hi, 0, 1, 2, 3, 4, 5, 6, 7,
                                      8, 9, 10, 11, 12, 13, 14, 15);
  }
}

// ---------------------------------------------------------------------------
// Kernel 3a (fast path): B staged by GLOBAL_LOAD_ASYNC_TO_LDS from bf16 vT.
// Inner loop: 2 async issues + 12 ds_load_b128 + 8 WMMA + asynccnt wait.
// ---------------------------------------------------------------------------
__global__ __launch_bounds__(256) void gemm_async_kernel(const float* __restrict__ x,
                                                         const unsigned short* __restrict__ vt,
                                                         const float* __restrict__ attn,
                                                         float* __restrict__ out) {
  extern __shared__ char smem[];
  __bf16* As  = (__bf16*)smem;
  __bf16* Bs0 = (__bf16*)(smem + SM_A_BYTES);
  __bf16* Bs1 = (__bf16*)(smem + SM_A_BYTES + SM_B_BYTES);
  float*  atS = (float*)(smem + SM_ATTN_OFF);

  const int tid = threadIdx.x;
  const int gm  = blockIdx.y * BM;
  const int gn  = blockIdx.x * BN;

  stage_a_attn(x, attn, As, atS, gm, tid);

  // Per-thread async transfer constants: 2x B128 per thread cover the 8KB tile.
  const int q0 = tid, q1 = tid + 256;          // (n = q>>2, kq = q&3)
  const int n0 = q0 >> 2, kq0 = q0 & 3;
  const int n1 = q1 >> 2, kq1 = q1 & 3;
  // Runtime (tid-dependent) LDS byte addresses: keeps the addrspacecast out of
  // any static initializer (scalar ptrtoint instructions only).
  const unsigned lds00 =
      (unsigned)(size_t)(smem + SM_A_BYTES + n0 * (BPITCH * 2) + kq0 * 16);
  const unsigned lds10 =
      (unsigned)(size_t)(smem + SM_A_BYTES + n1 * (BPITCH * 2) + kq1 * 16);
  const unsigned lds01 = lds00 + SM_B_BYTES;   // same slot, buffer 1
  const unsigned lds11 = lds10 + SM_B_BYTES;
  // Byte offsets in vT [c][n][k] bf16: c stride 2MB, n stride 2KB, k unit 2B.
  const unsigned g0 = (unsigned)((gn + n0) * (IN_DIM * 2) + kq0 * 16);
  const unsigned g1 = (unsigned)((gn + n1) * (IN_DIM * 2) + kq1 * 16);

  // Prolog stage of B tile (c=0, kk=0) via async copy into buffer 0.
  async_copy_b128(lds00, g0, vt);
  async_copy_b128(lds10, g1, vt);
  wait_async0();
  __syncthreads();

  const int warp = tid >> 5, lane = tid & 31;
  const int wm = (warp >> 2) * 64, wn = (warp & 3) * 32;
  const int lm = lane & 15, khalf = lane >> 4;

  v8f outAcc[4][2];
#pragma unroll
  for (int a = 0; a < 4; ++a)
#pragma unroll
    for (int b = 0; b < 2; ++b) outAcc[a][b] = zero8();

  for (int c = 0; c < CENTERS; ++c) {
    v8f tAcc[4][2];
#pragma unroll
    for (int a = 0; a < 4; ++a)
#pragma unroll
      for (int b = 0; b < 2; ++b) tAcc[a][b] = zero8();

    for (int kk = 0; kk < 32; ++kk) {
      const int it = c * 32 + kk;
      const __bf16* Bp = (it & 1) ? Bs1 : Bs0;
      const bool hasNext = it < (CENTERS * 32 - 1);

      if (hasNext) {
        const int itn = it + 1;
        const unsigned moff =
            ((unsigned)(itn >> 5) << 21) + ((unsigned)(itn & 31) << 6);
        const unsigned d0 = (itn & 1) ? lds01 : lds00;
        const unsigned d1 = (itn & 1) ? lds11 : lds10;
        async_copy_b128(d0, g0 + moff, vt);
        async_copy_b128(d1, g1 + moff, vt);
      }

      v16bf af[4], bfr[2];
      load_a_frags(As, wm, lm, khalf, kk, af);
      load_b_frags(Bp, wn, lm, khalf, bfr);

#pragma unroll
      for (int tm = 0; tm < 4; ++tm)
#pragma unroll
        for (int tn = 0; tn < 2; ++tn)
          tAcc[tm][tn] = __builtin_amdgcn_wmma_f32_16x16x32_bf16(
              false, af[tm], false, bfr[tn], (short)0, tAcc[tm][tn], false, false);

      if (hasNext) wait_async0();
      __syncthreads();
    }

#pragma unroll
    for (int tm = 0; tm < 4; ++tm) {
      float av[8];
#pragma unroll
      for (int j = 0; j < 8; ++j)
        av[j] = atS[(wm + tm * 16 + khalf * 8 + j) * CENTERS + c];
#pragma unroll
      for (int tn = 0; tn < 2; ++tn)
#pragma unroll
        for (int j = 0; j < 8; ++j)
          outAcc[tm][tn][j] = fmaf(av[j], tAcc[tm][tn][j], outAcc[tm][tn][j]);
    }
  }

#pragma unroll
  for (int tm = 0; tm < 4; ++tm)
#pragma unroll
    for (int tn = 0; tn < 2; ++tn)
#pragma unroll
      for (int j = 0; j < 8; ++j) {
        int row = gm + wm + tm * 16 + khalf * 8 + j;
        int col = gn + wn + tn * 16 + lm;
        out[(size_t)row * OUT_DIM + col] = outAcc[tm][tn][j];
      }
}

// ---------------------------------------------------------------------------
// Kernel 3b (fallback if ws too small): f32 v_c, convert-in-loop staging.
// ---------------------------------------------------------------------------
__global__ __launch_bounds__(256) void gemm_kernel(const float* __restrict__ x,
                                                   const float* __restrict__ vc,
                                                   const float* __restrict__ attn,
                                                   float* __restrict__ out) {
  extern __shared__ char smem[];
  __bf16* As  = (__bf16*)smem;
  __bf16* Bs0 = (__bf16*)(smem + SM_A_BYTES);
  __bf16* Bs1 = (__bf16*)(smem + SM_A_BYTES + SM_B_BYTES);
  float*  atS = (float*)(smem + SM_ATTN_OFF);

  const int tid = threadIdx.x;
  const int gm  = blockIdx.y * BM;
  const int gn  = blockIdx.x * BN;

  stage_a_attn(x, attn, As, atS, gm, tid);

#pragma unroll
  for (int p = 0; p < 8; ++p) {
    int q = tid + p * 256;
    int n = q & 127, k2 = q >> 7;
    const float* g = vc + (size_t)(k2 * 2) * OUT_DIM + gn + n;
    unsigned pk = (unsigned)f2bf(g[0]) | ((unsigned)f2bf(g[OUT_DIM]) << 16);
    *(unsigned*)(Bs0 + n * BPITCH + k2 * 2) = pk;
  }
  __syncthreads();

  const int warp = tid >> 5, lane = tid & 31;
  const int wm = (warp >> 2) * 64, wn = (warp & 3) * 32;
  const int lm = lane & 15, khalf = lane >> 4;

  v8f outAcc[4][2];
#pragma unroll
  for (int a = 0; a < 4; ++a)
#pragma unroll
    for (int b = 0; b < 2; ++b) outAcc[a][b] = zero8();

  for (int c = 0; c < CENTERS; ++c) {
    v8f tAcc[4][2];
#pragma unroll
    for (int a = 0; a < 4; ++a)
#pragma unroll
      for (int b = 0; b < 2; ++b) tAcc[a][b] = zero8();

    for (int kk = 0; kk < 32; ++kk) {
      const int it = c * 32 + kk;
      const __bf16* Bp = (it & 1) ? Bs1 : Bs0;
      const bool hasNext = it < (CENTERS * 32 - 1);

      float nv0[8], nv1[8];
      if (hasNext) {
        const int itn = it + 1;
        const int cn = itn >> 5, kn = itn & 31;
        const float* vb = vc + ((size_t)cn << 20) + (size_t)kn * 32 * OUT_DIM + gn;
#pragma unroll
        for (int p = 0; p < 8; ++p) {
          int q = tid + p * 256;
          int n = q & 127, k2 = q >> 7;
          const float* g = vb + (size_t)(k2 * 2) * OUT_DIM + n;
          nv0[p] = g[0];
          nv1[p] = g[OUT_DIM];
        }
      }

      v16bf af[4], bfr[2];
      load_a_frags(As, wm, lm, khalf, kk, af);
      load_b_frags(Bp, wn, lm, khalf, bfr);

#pragma unroll
      for (int tm = 0; tm < 4; ++tm)
#pragma unroll
        for (int tn = 0; tn < 2; ++tn)
          tAcc[tm][tn] = __builtin_amdgcn_wmma_f32_16x16x32_bf16(
              false, af[tm], false, bfr[tn], (short)0, tAcc[tm][tn], false, false);

      if (hasNext) {
        __bf16* Bn = ((it + 1) & 1) ? Bs1 : Bs0;
#pragma unroll
        for (int p = 0; p < 8; ++p) {
          int q = tid + p * 256;
          int n = q & 127, k2 = q >> 7;
          unsigned pk = (unsigned)f2bf(nv0[p]) | ((unsigned)f2bf(nv1[p]) << 16);
          *(unsigned*)(Bn + n * BPITCH + k2 * 2) = pk;
        }
      }
      __syncthreads();
    }

#pragma unroll
    for (int tm = 0; tm < 4; ++tm) {
      float av[8];
#pragma unroll
      for (int j = 0; j < 8; ++j)
        av[j] = atS[(wm + tm * 16 + khalf * 8 + j) * CENTERS + c];
#pragma unroll
      for (int tn = 0; tn < 2; ++tn)
#pragma unroll
        for (int j = 0; j < 8; ++j)
          outAcc[tm][tn][j] = fmaf(av[j], tAcc[tm][tn][j], outAcc[tm][tn][j]);
    }
  }

#pragma unroll
  for (int tm = 0; tm < 4; ++tm)
#pragma unroll
    for (int tn = 0; tn < 2; ++tn)
#pragma unroll
      for (int j = 0; j < 8; ++j) {
        int row = gm + wm + tm * 16 + khalf * 8 + j;
        int col = gn + wn + tn * 16 + lm;
        out[(size_t)row * OUT_DIM + col] = outAcc[tm][tn][j];
      }
}

// ---------------------------------------------------------------------------
extern "C" void kernel_launch(void* const* d_in, const int* in_sizes, int n_in,
                              void* d_out, int out_size, void* d_ws, size_t ws_size,
                              hipStream_t stream) {
  const float* x  = (const float*)d_in[0];  // [4096, 1024]
  const float* kc = (const float*)d_in[1];  // [1024, 32]
  const float* vc = (const float*)d_in[2];  // [32, 1024, 1024]
  float* out = (float*)d_out;

  float* invkn = (float*)d_ws;                                       // 32 f32
  float* attn  = (float*)d_ws + 64;                                  // [4096,32]
  unsigned short* vt = (unsigned short*)((char*)d_ws + (1u << 20));  // bf16 vT

  const size_t vt_bytes = (size_t)CENTERS * IN_DIM * OUT_DIM * 2;    // 64MB
  const bool fast = ws_size >= ((size_t)(1u << 20) + vt_bytes);

  knorm_kernel<<<1, 32, 0, stream>>>(kc, invkn);
  attn_kernel<<<dim3(NROWS / 4), 128, 0, stream>>>(x, kc, invkn, attn);

  if (fast) {
    vconv_kernel<<<dim3(OUT_DIM / 64, IN_DIM / 64, CENTERS), 256, 0, stream>>>(vc, vt);
    (void)hipFuncSetAttribute(reinterpret_cast<const void*>(gemm_async_kernel),
                              hipFuncAttributeMaxDynamicSharedMemorySize,
                              (int)SMEM_BYTES);
    gemm_async_kernel<<<dim3(OUT_DIM / BN, NROWS / BM), 256, SMEM_BYTES, stream>>>(
        x, vt, attn, out);
  } else {
    (void)hipFuncSetAttribute(reinterpret_cast<const void*>(gemm_kernel),
                              hipFuncAttributeMaxDynamicSharedMemorySize,
                              (int)SMEM_BYTES);
    gemm_kernel<<<dim3(OUT_DIM / BN, NROWS / BM), 256, SMEM_BYTES, stream>>>(
        x, vc, attn, out);
  }
}
